// Int4FusedLinear_6451040879136
// MI455X (gfx1250) — matrix operand order repaired and checked by reading, not compiled, not run on hardware
//
#include <hip/hip_runtime.h>

// Problem constants (from reference)
#define M_DIM 8192
#define K_DIM 4096
#define N4    8192
#define N8    2816
#define NT    11008

// Tiling
#define BM 128
#define BN 128
#define BK 32
#define LDS_STRIDE 40   // bf16 elems per LDS row (32 + 8 pad, keeps 16B alignment)

typedef __attribute__((ext_vector_type(16))) __bf16 v16bf;
typedef __attribute__((ext_vector_type(8)))  float  v8f;

__device__ __forceinline__ unsigned short f32_bf16_rne(float f) {
  unsigned u = __float_as_uint(f);
  u += 0x7FFFu + ((u >> 16) & 1u);          // round-to-nearest-even
  return (unsigned short)(u >> 16);
}
__device__ __forceinline__ float bf16_f32(unsigned short h) {
  return __uint_as_float(((unsigned)h) << 16);
}
__device__ __forceinline__ void split4(const float4 v, ushort4& h, ushort4& l) {
  h.x = f32_bf16_rne(v.x); l.x = f32_bf16_rne(v.x - bf16_f32(h.x));
  h.y = f32_bf16_rne(v.y); l.y = f32_bf16_rne(v.y - bf16_f32(h.y));
  h.z = f32_bf16_rne(v.z); l.z = f32_bf16_rne(v.z - bf16_f32(h.z));
  h.w = f32_bf16_rne(v.w); l.w = f32_bf16_rne(v.w - bf16_f32(h.w));
}

union FragU { v16bf v; uint4 q[2]; };

// A fragment (16x32 bf16, MxK). Contiguous k order (same as B):
//   lane L: M = L&15 ; elements e -> K = 16*(L>>4) + e
__device__ __forceinline__ v16bf load_frag_a(const unsigned short* __restrict__ s,
                                             int mBase, int lane) {
  const int m  = mBase + (lane & 15);
  const int k0 = (lane >> 4) << 4;
  const unsigned short* row = s + m * LDS_STRIDE;
  FragU u;
  u.q[0] = *(const uint4*)(row + k0);
  u.q[1] = *(const uint4*)(row + k0 + 8);
  return u.v;
}

// B fragment (32x16 bf16, KxN), B[k][n] = Wperm[n][k]:
//   lane L: N = L&15 ; elements e -> K = 16*(L>>4) + e
__device__ __forceinline__ v16bf load_frag_b(const unsigned short* __restrict__ s,
                                             int nBase, int lane) {
  const int n  = nBase + (lane & 15);
  const int k0 = (lane >> 4) << 4;
  const unsigned short* row = s + n * LDS_STRIDE;
  FragU u;
  u.q[0] = *(const uint4*)(row + k0);
  u.q[1] = *(const uint4*)(row + k0 + 8);
  return u.v;
}

// CDNA5 async global->LDS copy (16B per lane), tracked by ASYNCcnt.
#define ASYNC_LD16(ldsoff, gaddr, OFFSTR)                                   \
  asm volatile("global_load_async_to_lds_b128 %0, %1, off offset:" OFFSTR   \
               :: "v"(ldsoff), "v"(gaddr) : "memory")
__device__ __forceinline__ void wait_async0() {
  asm volatile("s_wait_asynccnt 0x0" ::: "memory");
}
__device__ __forceinline__ unsigned lds_off_of(const void* p) {
  return (unsigned)(unsigned long long)(uintptr_t)p;   // low 32 bits of generic addr = LDS offset
}

// ---------------------------------------------------------------------------
// Pass 1a: split x (fp32) into bf16 hi/lo, done ONCE (vs once per N-block).
// ---------------------------------------------------------------------------
__global__ __launch_bounds__(256)
void convert_split_x(const float* __restrict__ src,
                     unsigned short* __restrict__ hi,
                     unsigned short* __restrict__ lo) {
  const size_t idx = ((size_t)blockIdx.x * 256 + threadIdx.x) * 4;
  const float4 v = *(const float4*)(src + idx);
  ushort4 h, l; split4(v, h, l);
  *(ushort4*)(hi + idx) = h;
  *(ushort4*)(lo + idx) = l;
}

// ---------------------------------------------------------------------------
// Pass 1b: gather W rows through inv_perm (fuses jnp.take) + split to bf16.
// ---------------------------------------------------------------------------
__global__ __launch_bounds__(256)
void gather_split_w(const float* __restrict__ w4, const float* __restrict__ w8,
                    const long long* __restrict__ inv_perm,
                    unsigned short* __restrict__ hi,
                    unsigned short* __restrict__ lo) {
  const int j = blockIdx.y;
  const long long n = inv_perm[j];
  const float* src = (n < (long long)N4) ? (w4 + (size_t)n * K_DIM)
                                         : (w8 + (size_t)(n - N4) * K_DIM);
  const int c = (blockIdx.x * 256 + threadIdx.x) * 4;
  const float4 v = *(const float4*)(src + c);
  ushort4 h, l; split4(v, h, l);
  const size_t o = (size_t)j * K_DIM + c;
  *(ushort4*)(hi + o) = h;
  *(ushort4*)(lo + o) = l;
}

// ---------------------------------------------------------------------------
// Pass 2: bf16 split-GEMM. Double-buffered async global->LDS staging so the
// ASYNCcnt copies overlap the 24 WMMAs of the live stage.
// ---------------------------------------------------------------------------
__device__ __forceinline__ void compute_stage(const unsigned short* __restrict__ Ah,
                                              const unsigned short* __restrict__ Al,
                                              const unsigned short* __restrict__ Bh,
                                              const unsigned short* __restrict__ Bl,
                                              int wm, int wn, int lane,
                                              v8f (&acc)[2][4]) {
  const int mA0 = wm * 32;
  v16bf a_hi[2], a_lo[2];
#pragma unroll
  for (int i = 0; i < 2; ++i) {
    a_hi[i] = load_frag_a(Ah, mA0 + i * 16, lane);
    a_lo[i] = load_frag_a(Al, mA0 + i * 16, lane);
  }
#pragma unroll
  for (int j = 0; j < 4; ++j) {
    const v16bf b_hi = load_frag_b(Bh, wn * 64 + j * 16, lane);
    const v16bf b_lo = load_frag_b(Bl, wn * 64 + j * 16, lane);
#pragma unroll
    for (int i = 0; i < 2; ++i) {
      acc[i][j] = __builtin_amdgcn_wmma_f32_16x16x32_bf16(
          false, a_hi[i], false, b_hi, (short)0, acc[i][j], false, false);
      acc[i][j] = __builtin_amdgcn_wmma_f32_16x16x32_bf16(
          false, a_lo[i], false, b_hi, (short)0, acc[i][j], false, false);
      acc[i][j] = __builtin_amdgcn_wmma_f32_16x16x32_bf16(
          false, a_hi[i], false, b_lo, (short)0, acc[i][j], false, false);
    }
  }
}

__global__ __launch_bounds__(256)
void gemm_bf16_split(const unsigned short* __restrict__ xh,
                     const unsigned short* __restrict__ xl,
                     const unsigned short* __restrict__ wh,
                     const unsigned short* __restrict__ wl,
                     const float* __restrict__ bias,
                     float* __restrict__ out) {
  __shared__ __align__(16) unsigned short As_hi[2][BM * LDS_STRIDE];
  __shared__ __align__(16) unsigned short As_lo[2][BM * LDS_STRIDE];
  __shared__ __align__(16) unsigned short Bs_hi[2][BN * LDS_STRIDE];
  __shared__ __align__(16) unsigned short Bs_lo[2][BN * LDS_STRIDE];

  const int tid  = threadIdx.x;
  const int lane = tid & 31;
  const int wid  = tid >> 5;
  const int wm   = wid >> 1;     // 0..3 : 32-row strip
  const int wn   = wid & 1;      // 0..1 : 64-col strip

  const int bn0 = blockIdx.x * BN;
  const int bm0 = blockIdx.y * BM;

  // Staging slots: thread covers rows r and r+64 at 16B segment seg.
  const int r   = tid >> 2;      // 0..63
  const int seg = tid & 3;       // 0..3 (x16B)

  // Global byte addresses (advance by 2*BK*2 = 128 bytes per unrolled pair;
  // the odd stage uses the instruction's immediate offset:64).
  unsigned long long gAh0 = (unsigned long long)(uintptr_t)(xh + (size_t)(bm0 + r)      * K_DIM + seg * 8);
  unsigned long long gAh1 = (unsigned long long)(uintptr_t)(xh + (size_t)(bm0 + r + 64) * K_DIM + seg * 8);
  unsigned long long gAl0 = (unsigned long long)(uintptr_t)(xl + (size_t)(bm0 + r)      * K_DIM + seg * 8);
  unsigned long long gAl1 = (unsigned long long)(uintptr_t)(xl + (size_t)(bm0 + r + 64) * K_DIM + seg * 8);
  unsigned long long gBh0 = (unsigned long long)(uintptr_t)(wh + (size_t)(bn0 + r)      * K_DIM + seg * 8);
  unsigned long long gBh1 = (unsigned long long)(uintptr_t)(wh + (size_t)(bn0 + r + 64) * K_DIM + seg * 8);
  unsigned long long gBl0 = (unsigned long long)(uintptr_t)(wl + (size_t)(bn0 + r)      * K_DIM + seg * 8);
  unsigned long long gBl1 = (unsigned long long)(uintptr_t)(wl + (size_t)(bn0 + r + 64) * K_DIM + seg * 8);

  // LDS byte offsets.
  const unsigned ldsRow0 = (unsigned)((r      * LDS_STRIDE + seg * 8) * 2);
  const unsigned ldsRow1 = (unsigned)(((r+64) * LDS_STRIDE + seg * 8) * 2);
  const unsigned lAh[2] = { lds_off_of(As_hi[0]), lds_off_of(As_hi[1]) };
  const unsigned lAl[2] = { lds_off_of(As_lo[0]), lds_off_of(As_lo[1]) };
  const unsigned lBh[2] = { lds_off_of(Bs_hi[0]), lds_off_of(Bs_hi[1]) };
  const unsigned lBl[2] = { lds_off_of(Bs_lo[0]), lds_off_of(Bs_lo[1]) };

#define ISSUE_GROUP(B, OFFSTR) do {                                          \
    ASYNC_LD16(lAh[B] + ldsRow0, gAh0, OFFSTR);                              \
    ASYNC_LD16(lAh[B] + ldsRow1, gAh1, OFFSTR);                              \
    ASYNC_LD16(lAl[B] + ldsRow0, gAl0, OFFSTR);                              \
    ASYNC_LD16(lAl[B] + ldsRow1, gAl1, OFFSTR);                              \
    ASYNC_LD16(lBh[B] + ldsRow0, gBh0, OFFSTR);                              \
    ASYNC_LD16(lBh[B] + ldsRow1, gBh1, OFFSTR);                              \
    ASYNC_LD16(lBl[B] + ldsRow0, gBl0, OFFSTR);                              \
    ASYNC_LD16(lBl[B] + ldsRow1, gBl1, OFFSTR);                              \
  } while (0)

  v8f acc[2][4] = {};

  // Prologue: stage 0 -> buf0.
  ISSUE_GROUP(0, "0");

  for (int kk = 0; kk < K_DIM; kk += 2 * BK) {
    // ---- stage kk : compute buf0, prefetch kk+BK -> buf1 ----
    wait_async0();
    __syncthreads();                    // buf0 landed; all waves done reading buf1
    if (kk + BK < K_DIM) ISSUE_GROUP(1, "64");
    compute_stage(As_hi[0], As_lo[0], Bs_hi[0], Bs_lo[0], wm, wn, lane, acc);

    // ---- stage kk+BK : compute buf1, prefetch kk+2*BK -> buf0 ----
    wait_async0();
    __syncthreads();                    // buf1 landed; all waves done reading buf0
    gAh0 += 128; gAh1 += 128; gAl0 += 128; gAl1 += 128;
    gBh0 += 128; gBh1 += 128; gBl0 += 128; gBl1 += 128;
    if (kk + 2 * BK < K_DIM) ISSUE_GROUP(0, "0");
    compute_stage(As_hi[1], As_lo[1], Bs_hi[1], Bs_lo[1], wm, wn, lane, acc);
  }
#undef ISSUE_GROUP

  // ---- Epilogue: bias add, store (permutation already applied in pass 1b) ----
  const int colLane = lane & 15;
  const int rowHalf = (lane >> 4) << 3;
#pragma unroll
  for (int i = 0; i < 2; ++i) {
    const int mg = bm0 + wm * 32 + i * 16 + rowHalf;
#pragma unroll
    for (int j = 0; j < 4; ++j) {
      const int ng = bn0 + wn * 64 + j * 16 + colLane;
      const float bv = bias[ng];
      float* o = out + (size_t)mg * NT + ng;
#pragma unroll
      for (int rr = 0; rr < 8; ++rr)
        o[(size_t)rr * NT] = acc[i][j][rr] + bv;
    }
  }
}

// ---------------------------------------------------------------------------
// Fallback: fused single-pass kernel (used when ws is too small).
// ---------------------------------------------------------------------------
__global__ __launch_bounds__(256)
void fused_gemm_perm_bias(const float* __restrict__ x,
                          const float* __restrict__ w4,
                          const float* __restrict__ w8,
                          const long long* __restrict__ inv_perm,
                          const float* __restrict__ bias,
                          float* __restrict__ out) {
  __shared__ __align__(16) unsigned short As_hi[BM * LDS_STRIDE];
  __shared__ __align__(16) unsigned short As_lo[BM * LDS_STRIDE];
  __shared__ __align__(16) unsigned short Bs_hi[BN * LDS_STRIDE];
  __shared__ __align__(16) unsigned short Bs_lo[BN * LDS_STRIDE];
  __shared__ const float* rowPtr[BN];

  const int tid  = threadIdx.x;
  const int lane = tid & 31;
  const int wid  = tid >> 5;
  const int wm   = wid >> 1;
  const int wn   = wid & 1;
  const int bn0  = blockIdx.x * BN;
  const int bm0  = blockIdx.y * BM;

  if (tid < BN) {
    const long long n = inv_perm[bn0 + tid];
    rowPtr[tid] = (n < (long long)N4) ? (w4 + (size_t)n * K_DIM)
                                      : (w8 + (size_t)(n - N4) * K_DIM);
  }
  __syncthreads();

  v8f acc[2][4] = {};

  for (int kk = 0; kk < K_DIM; kk += BK) {
#pragma unroll
    for (int it = 0; it < 4; ++it) {
      const int s = tid + it * 256;
      const int m = s >> 3;
      const int c = (s & 7) << 2;
      const float4 v = *(const float4*)(x + (size_t)(bm0 + m) * K_DIM + kk + c);
      ushort4 h, l; split4(v, h, l);
      *(ushort4*)(As_hi + m * LDS_STRIDE + c) = h;
      *(ushort4*)(As_lo + m * LDS_STRIDE + c) = l;
    }
#pragma unroll
    for (int it = 0; it < 4; ++it) {
      const int s = tid + it * 256;
      const int n = s >> 3;
      const int c = (s & 7) << 2;
      const float4 v = *(const float4*)(rowPtr[n] + kk + c);
      ushort4 h, l; split4(v, h, l);
      *(ushort4*)(Bs_hi + n * LDS_STRIDE + c) = h;
      *(ushort4*)(Bs_lo + n * LDS_STRIDE + c) = l;
    }
    __syncthreads();
    compute_stage(As_hi, As_lo, Bs_hi, Bs_lo, wm, wn, lane, acc);
    __syncthreads();
  }

  const int colLane = lane & 15;
  const int rowHalf = (lane >> 4) << 3;
#pragma unroll
  for (int i = 0; i < 2; ++i) {
    const int mg = bm0 + wm * 32 + i * 16 + rowHalf;
#pragma unroll
    for (int j = 0; j < 4; ++j) {
      const int ng = bn0 + wn * 64 + j * 16 + colLane;
      const float bv = bias[ng];
      float* o = out + (size_t)mg * NT + ng;
#pragma unroll
      for (int rr = 0; rr < 8; ++rr)
        o[(size_t)rr * NT] = acc[i][j][rr] + bv;
    }
  }
}

extern "C" void kernel_launch(void* const* d_in, const int* in_sizes, int n_in,
                              void* d_out, int out_size, void* d_ws, size_t ws_size,
                              hipStream_t stream) {
  const float*     x    = (const float*)d_in[0];
  const float*     w4   = (const float*)d_in[1];
  const float*     w8   = (const float*)d_in[2];
  const long long* ip   = (const long long*)d_in[3];
  const float*     bias = (const float*)d_in[4];
  float*           out  = (float*)d_out;

  const size_t xElems = (size_t)M_DIM * K_DIM;   // 33.5M
  const size_t wElems = (size_t)NT * K_DIM;      // 45.1M
  const size_t need   = (xElems + wElems) * 2 * sizeof(unsigned short); // ~300 MB

  if (ws_size >= need) {
    unsigned short* xh = (unsigned short*)d_ws;
    unsigned short* xl = xh + xElems;
    unsigned short* wh = xl + xElems;
    unsigned short* wl = wh + wElems;

    convert_split_x<<<dim3((unsigned)(xElems / 1024)), dim3(256), 0, stream>>>(x, xh, xl);
    gather_split_w<<<dim3(K_DIM / 1024, NT), dim3(256), 0, stream>>>(w4, w8, ip, wh, wl);
    gemm_bf16_split<<<dim3(NT / BN, M_DIM / BM), dim3(256), 0, stream>>>(xh, xl, wh, wl, bias, out);
  } else {
    fused_gemm_perm_bias<<<dim3(NT / BN, M_DIM / BM), dim3(256), 0, stream>>>(x, w4, w8, ip, bias, out);
  }
}
